// STDP_14877766713533
// MI455X (gfx1250) — matrix operand order repaired and checked by reading, not compiled, not run on hardware
//
#include <hip/hip_runtime.h>

// ---------------------------------------------------------------------------
// STDP weight update for MI455X (gfx1250, wave32, WMMA + TDM).
//
// dW[i,j] = sum_{b,q} M[b,q,i] * post[b,q,j], M = exponential STDP traces of
// pre (forward + backward scans, O(T) instead of the reference's O(T^2)),
// out = weights + dW via a K=16384 f16 WMMA GEMM with f32 accumulation.
//
// v4: same TDM double-buffered pipeline as v3, plus an escape of the LDS
// arena into an asm-with-memory-clobber: the Tensor Data Mover writes LDS
// behind the compiler's back (the descriptor carries the LDS address only as
// an integer), and without the escape alias analysis proved smem "never
// written" and deleted the WMMA fragment loads entirely (seen in v3 asm).
//
// Workspace (64 MB):
//   [0, 32MB)   : MhT = f16 trace matrix, layout [i][k], k = b*T + q
//   [32MB,64MB) : PhT = f16 post spikes,  layout [j][k]
// ---------------------------------------------------------------------------

typedef __attribute__((ext_vector_type(16))) _Float16 v16h;
typedef __attribute__((ext_vector_type(8)))  _Float16 v8h;
typedef __attribute__((ext_vector_type(4)))  _Float16 v4h;
typedef __attribute__((ext_vector_type(8)))  float    v8f;
typedef __attribute__((ext_vector_type(4)))  float    v4f;
typedef __attribute__((ext_vector_type(4)))  unsigned int u32x4;
typedef __attribute__((ext_vector_type(8)))  int      i32x8;
typedef __attribute__((ext_vector_type(4)))  int      i32x4;

#define N_PRE   1024
#define N_POST  1024
#define T_STEPS 2048
#define BATCH   8
#define K_TOT   (BATCH * T_STEPS)      // 16384

#define A_PRE_C  0.01f
#define A_POST_C 0.01f
#define DECAY    0.951229424500714f    // exp(-1/20), TAU_PRE == TAU_POST == 20

// LDS arena layout (halves). Row stride 40 halves = 80 B (16B multiple).
#define LMS      40
#define LM_HALVES (128 * LMS)          // one A buffer: 128 rows x 32 k (padded)
#define LP_HALVES (64 * LMS)           // one B buffer: 64 rows x 32 k (padded)
#define LM_OFF(p) ((p) * LM_HALVES)
#define LP_OFF(p) (2 * LM_HALVES + (p) * LP_HALVES)

#if defined(__gfx1250__) && __has_builtin(__builtin_amdgcn_tensor_load_to_lds)
#define USE_TDM 1
#else
#define USE_TDM 0
#endif

// ---------------------------------------------------------------------------
// Kernel 1: exponential STDP traces -> MhT[i][k] (k-contiguous rows).
// ---------------------------------------------------------------------------
__global__ void __launch_bounds__(256)
stdp_trace_kernel(const float* __restrict__ pre, _Float16* __restrict__ MhT) {
    const int tid = blockIdx.x * blockDim.x + threadIdx.x;   // 0 .. B*N_PRE-1
    const int b = tid >> 10;
    const int i = tid & (N_PRE - 1);

    const float* prebi = pre + (size_t)b * T_STEPS * N_PRE + i;     // stride N_PRE
    _Float16*    mrow  = MhT + (size_t)i * K_TOT + (size_t)b * T_STEPS;

    // Forward: M_pot[q] = A_POST * sum_{p<q} pre[p] * exp(-(q-p)/tau)
    float S = 0.0f;
    for (int qc = 0; qc < T_STEPS; qc += 8) {
        v8h chunk;
#pragma unroll
        for (int e = 0; e < 8; ++e) {
            chunk[e] = (_Float16)(A_POST_C * S);
            S = DECAY * (S + prebi[(size_t)(qc + e) * N_PRE]);
        }
        *(v8h*)(mrow + qc) = chunk;
    }
    // Backward: M[q] -= A_PRE * sum_{p>q} pre[p] * exp(-(p-q)/tau)
    S = 0.0f;
    for (int qc = T_STEPS - 8; qc >= 0; qc -= 8) {
        v8h chunk = *(v8h*)(mrow + qc);
#pragma unroll
        for (int e = 7; e >= 0; --e) {
            chunk[e] = (_Float16)((float)chunk[e] - A_PRE_C * S);
            S = DECAY * (S + prebi[(size_t)(qc + e) * N_PRE]);
        }
        *(v8h*)(mrow + qc) = chunk;
    }
}

// ---------------------------------------------------------------------------
// Kernel 2: transpose + f32->f16 re-encode: post[k][j] -> PhT[j][k].
// ---------------------------------------------------------------------------
__global__ void __launch_bounds__(256)
cvt_post_kernel(const float* __restrict__ post, _Float16* __restrict__ PhT) {
    __shared__ __align__(16) _Float16 lt[32][36];   // [j][k], padded

    const int t     = threadIdx.x;
    const int kbase = blockIdx.x * 32;
    const int jbase = blockIdx.y * 32;
    {
        const int k  = t >> 3;
        const int j4 = (t & 7) * 4;
        v4f p = *(const v4f*)(post + (size_t)(kbase + k) * N_POST + jbase + j4);
#pragma unroll
        for (int u = 0; u < 4; ++u) lt[j4 + u][k] = (_Float16)p[u];
    }
    __syncthreads();
    {
        const int j  = t >> 3;
        const int kc = (t & 7) * 4;
        v4h h = *(const v4h*)&lt[j][kc];
        *(v4h*)(PhT + (size_t)(jbase + j) * K_TOT + kbase + kc) = h;
    }
}

// ---------------------------------------------------------------------------
// TDM descriptor issue: 2-D tile (tile_rows x 32 halves) of a [rows][K_TOT]
// f16 matrix -> LDS at byte address lds_addr, padding 16 B after every 64 B
// row (matches LMS=40-half LDS stride). Fields per CDNA5 ISA.
// ---------------------------------------------------------------------------
#if USE_TDM
__device__ __forceinline__ void tdm_load_tile(unsigned int lds_addr,
                                              const _Float16* gsrc,
                                              unsigned int tile_rows) {
    unsigned long long ga = (unsigned long long)gsrc;

    u32x4 g0;
    g0[0] = 1u;                                   // count=1 (valid user D#)
    g0[1] = lds_addr;                             // LDS byte address
    g0[2] = (unsigned int)(ga & 0xFFFFFFFFu);     // global_addr[31:0]
    g0[3] = (unsigned int)((ga >> 32) & 0x01FFFFFFu) | (2u << 30); // [56:32], type=2

    // group1: data_size=1(2B); pad_enable; pad_interval=3 (16 DWORDs = one
    // 32-half row); pad_amount=3 (4 DWORDs = 16 B = 8 halves of padding)
    const unsigned int w0  = (1u << 16) | (1u << 20) | (3u << 22) | (3u << 25);
    const unsigned int td0 = (unsigned int)K_TOT;   // tensor_dim0 (k extent)
    const unsigned int td1 = tile_rows;             // tensor_dim1 (rows)
    const unsigned long long s0 = (unsigned long long)K_TOT; // row stride (elems)

    i32x8 g1;
    g1[0] = (int)w0;
    g1[1] = (int)((td0 & 0xFFFFu) << 16);                    // dim0[15:0] @63:48
    g1[2] = (int)((td0 >> 16) | ((td1 & 0xFFFFu) << 16));    // dim0[31:16], dim1[15:0]
    g1[3] = (int)((td1 >> 16) | (32u << 16));                // dim1[31:16], tile_dim0=32
    g1[4] = (int)(tile_rows & 0xFFFFu);                      // tile_dim1; tile_dim2=0
    g1[5] = (int)(s0 & 0xFFFFFFFFu);                         // dim0_stride[31:0]
    g1[6] = (int)((s0 >> 32) & 0xFFFFu);                     // dim0_stride[47:32]
    g1[7] = 0;                                               // dim1_stride (unused, 2D)

    i32x4 gz = {};
#if __has_include(<hip/amd_detail/amd_gfx1250_TDM.h>)
    i32x8 gz8 = {};
    __builtin_amdgcn_tensor_load_to_lds(g0, g1, gz, gz, gz8, 0);   // clang-23 form
#else
    __builtin_amdgcn_tensor_load_to_lds(g0, g1, gz, gz, 0);        // ROCm 7.2 form
#endif
}
#endif

// ---------------------------------------------------------------------------
// Kernel 3: out = W + MhT @ PhT^T  via v_wmma_f32_16x16x32_f16.
// Grid (16, 8) x 256 threads (8 waves); 128x64 output block per WG.
// TDM double-buffered pipeline: DMA of tile k+1 overlaps compute of tile k.
// ---------------------------------------------------------------------------
__global__ void __launch_bounds__(256)
stdp_wmma_gemm(const _Float16* __restrict__ MhT,
               const _Float16* __restrict__ PhT,
               const float*    __restrict__ W,
               float*          __restrict__ out) {
    // single arena => LDS byte offsets of the buffers are known constants
    __shared__ __align__(16) _Float16 smem[2 * LM_HALVES + 2 * LP_HALVES];

#if USE_TDM
    // Escape the arena: the TDM writes LDS out-of-band (descriptor holds the
    // LDS address as an integer), so without this the compiler proves smem is
    // never stored and folds all fragment loads to undef (observed in v3).
    {
        void* esc = (void*)smem;
        asm volatile("" : "+v"(esc) :: "memory");
    }
#endif

    const int t    = threadIdx.x;
    const int wave = t >> 5;
    const int lane = t & 31;
    const int m    = lane & 15;
    const int kh   = lane >> 4;

    const int ibase = blockIdx.y * 128;  // output row block (N_pre)
    const int jbase = blockIdx.x * 64;   // output col block (N_post)
    const int roff  = wave * 16;

    const _Float16* gmTile = MhT + (size_t)ibase * K_TOT;  // + kb per step
    const _Float16* gpTile = PhT + (size_t)jbase * K_TOT;

    v8f zero = {};
    v8f acc[4];
#pragma unroll
    for (int c = 0; c < 4; ++c) acc[c] = zero;

#if !USE_TDM
    // -------- fallback: synchronous b128 staging (v2 path) --------
    const int mrow = t >> 1, moff = (t & 1) * 16;
    const int prow = t >> 2, poff = (t & 3) * 8;
    const _Float16* gmBase = gmTile + (size_t)mrow * K_TOT + moff;
    const _Float16* gpBase = gpTile + (size_t)prow * K_TOT + poff;
#else
    // -------- TDM prologue: DMA first tile into buffer 0 --------
    if (t == 0) {
        tdm_load_tile(2u * LM_OFF(0), gmTile, 128u);
        tdm_load_tile(2u * LP_OFF(0), gpTile, 64u);
    }
#endif

#pragma unroll 2
    for (int it = 0; it < K_TOT / 32; ++it) {
        const int kb = it * 32;
        const int p  = it & 1;

#if USE_TDM
        if (t == 0) __builtin_amdgcn_s_wait_tensorcnt((short)0);  // tile `it` landed
        __syncthreads();   // data visible; everyone done reading the other buffer
        if (t == 0 && it + 1 < K_TOT / 32) {                      // DMA next tile
            tdm_load_tile(2u * LM_OFF(p ^ 1), gmTile + kb + 32, 128u);
            tdm_load_tile(2u * LP_OFF(p ^ 1), gpTile + kb + 32, 64u);
        }
        const _Float16* lmP = smem + LM_OFF(p);
        const _Float16* lpP = smem + LP_OFF(p);
#else
        __syncthreads();
        {
            const v8h* gm = (const v8h*)(gmBase + kb);
            *(v8h*)&smem[LM_OFF(0) + mrow * LMS + moff]     = gm[0];
            *(v8h*)&smem[LM_OFF(0) + mrow * LMS + moff + 8] = gm[1];
            *(v8h*)&smem[LP_OFF(0) + prow * LMS + poff]     = *(const v8h*)(gpBase + kb);
            __builtin_prefetch(gmBase + kb + 32, 0, 1);
            __builtin_prefetch(gpBase + kb + 32, 0, 1);
        }
        __syncthreads();
        const _Float16* lmP = smem + LM_OFF(0);
        const _Float16* lpP = smem + LP_OFF(0);
#endif

        // A fragment (ISA 7.12.2): elems 0..7 = A[m][kh*8..+8], 8..15 = A[m][16+kh*8..+8]
        v8h a0 = *(const v8h*)&lmP[(roff + m) * LMS + kh * 8];
        v8h a1 = *(const v8h*)&lmP[(roff + m) * LMS + 16 + kh * 8];
        v16h a = __builtin_shufflevector(a0, a1,
                    0, 1, 2, 3, 4, 5, 6, 7, 8, 9, 10, 11, 12, 13, 14, 15);

#pragma unroll
        for (int c = 0; c < 4; ++c) {
            // B fragment: lane column n = lane&15, elems 0..7 = K kh*8..+8, 8..15 = K 16+kh*8..+8
            v8h b0 = *(const v8h*)&lpP[(c * 16 + m) * LMS + kh * 8];
            v8h b1 = *(const v8h*)&lpP[(c * 16 + m) * LMS + 16 + kh * 8];
            v16h bb = __builtin_shufflevector(b0, b1,
                        0, 1, 2, 3, 4, 5, 6, 7, 8, 9, 10, 11, 12, 13, 14, 15);
            acc[c] = __builtin_amdgcn_wmma_f32_16x16x32_f16(
                false, a, false, bb, (short)0, acc[c], false, false);
        }
    }

    // Epilogue: D f32 layout -> row = roff + (lane>>4)*8 + r, col = lane&15
    const int mh = lane >> 4;
#pragma unroll
    for (int c = 0; c < 4; ++c) {
#pragma unroll
        for (int r = 0; r < 8; ++r) {
            const int row = ibase + roff + mh * 8 + r;
            const int col = jbase + (c << 4) + m;
            const size_t o = (size_t)row * N_POST + col;
            out[o] = W[o] + acc[c][r];
        }
    }
}

// ---------------------------------------------------------------------------
extern "C" void kernel_launch(void* const* d_in, const int* in_sizes, int n_in,
                              void* d_out, int out_size, void* d_ws, size_t ws_size,
                              hipStream_t stream) {
    (void)in_sizes; (void)n_in; (void)out_size; (void)ws_size;

    const float* W    = (const float*)d_in[0];   // [1024,1024] f32
    const float* pre  = (const float*)d_in[1];   // [8,2048,1024] f32
    const float* post = (const float*)d_in[2];   // [8,2048,1024] f32
    float*       out  = (float*)d_out;           // [1024,1024] f32

    _Float16* MhT = (_Float16*)d_ws;                                         // 32 MB
    _Float16* PhT = (_Float16*)((char*)d_ws + (size_t)N_PRE * K_TOT * 2);    // 32 MB

    stdp_trace_kernel<<<dim3((BATCH * N_PRE) / 256), dim3(256), 0, stream>>>(pre, MhT);
    cvt_post_kernel<<<dim3(K_TOT / 32, N_POST / 32), dim3(256), 0, stream>>>(post, PhT);
    stdp_wmma_gemm<<<dim3(N_POST / 64, N_PRE / 128), dim3(256), 0, stream>>>(MhT, PhT, W, out);
}
